// D4DispersionEnergy_81733227643308
// MI455X (gfx1250) — matrix operand order repaired and checked by reading, not compiled, not run on hardware
//
#include <hip/hip_runtime.h>
#include <math.h>

#define NREF 7
#define NC 7

typedef _Float16 v16h __attribute__((ext_vector_type(16)));
typedef float    v8f  __attribute__((ext_vector_type(8)));

__device__ __forceinline__ float softplusf(float x) { return log1pf(expf(x)); }

__device__ __forceinline__ float switchf(float d) {
    const float cuton = 1.875f, cutoff = 7.5f;
    float x = (d - cuton) / (cutoff - cuton);
    float x3 = x * x * x;
    float sw = 1.f - 6.f * x3 * x * x + 15.f * x3 * x - 10.f * x3;
    return (x <= 0.f) ? 1.f : ((x >= 1.f) ? 0.f : sw);
}

__device__ __forceinline__ void wait_asynccnt0() {
#if __has_builtin(__builtin_amdgcn_s_wait_asynccnt)
    __builtin_amdgcn_s_wait_asynccnt(0);
#else
    asm volatile("s_wait_asynccnt 0x0" ::: "memory");
#endif
}

// ---------------------------------------------------------------- init ------
__global__ void k_init(float* covcn, float* y_atom, float* y, int n, int nmol) {
    int t = blockIdx.x * blockDim.x + threadIdx.x;
    if (t < n) { covcn[t] = 0.f; y_atom[t] = 0.f; }
    if (t < nmol) y[t] = 0.f;
}

// ------------------------------------------------------- pass 1: CN ---------
__global__ void k_cn(const float* __restrict__ r_ij, const int* __restrict__ idx_i,
                     const int* __restrict__ idx_j, const int* __restrict__ Z,
                     const float* __restrict__ rcov, const float* __restrict__ en,
                     float* __restrict__ covcn, int P) {
    int p = blockIdx.x * blockDim.x + threadIdx.x;
    if (p >= P) return;
    int i = idx_i[p], j = idx_j[p];
    int Zi = Z[i], Zj = Z[j];
    float rx = r_ij[3 * p], ry = r_ij[3 * p + 1], rz = r_ij[3 * p + 2];
    float d = sqrtf(rx * rx + ry * ry + rz * rz);
    float rco = (4.f / 3.f) * (rcov[Zi] + rcov[Zj]);
    float de = fabsf(en[Zi] - en[Zj]) + 19.08857f;
    float den = 4.10451f * expf(-de * de / 254.5553148552f);
    float tmp = den * 0.5f * (1.f + erff(-7.5f * (d - rco) / rco));
    tmp *= switchf(d);
    atomicAdd(&covcn[i], tmp);
}

// ------------------------------------------------------ pass 2: zeta --------
__global__ void k_zeta(const int* __restrict__ Z, const float* __restrict__ covcn,
                       const float* __restrict__ qa, const float* __restrict__ ncm,
                       const float* __restrict__ ncw, const float* __restrict__ cn,
                       const float* __restrict__ fixg, const float* __restrict__ refq,
                       const float* __restrict__ zeff, const float* __restrict__ gam,
                       const float* __restrict__ scaleq_raw,
                       float* __restrict__ zeta, int n) {
    int a = blockIdx.x * blockDim.x + threadIdx.x;
    if (a >= n) return;
    int z = Z[a];
    float c = covcn[a];
    float g[NREF];
    float norm = 0.f;
#pragma unroll
    for (int r = 0; r < NREF; r++) {
        float s = 0.f;
        int base = (z * NREF + r) * NC;
#pragma unroll
        for (int k = 0; k < NC; k++) {
            float diff = c - cn[base + k];
            s += ncm[base + k] * expf(-6.f * ncw[base + k] * diff * diff);
        }
        g[r] = s;
        norm += s;
    }
    bool ok = norm > 1e-8f;
    float inv = ok ? 1.f / norm : 0.f;
    float scaleq = softplusf(scaleq_raw[0]);
    float iz = zeff[z];
    float qmod = iz + qa[a];
    bool qok = qmod > 1e-8f;
    float invq = qok ? 1.f / qmod : 0.f;
    float gz = gam[z];
#pragma unroll
    for (int r = 0; r < NREF; r++) {
        float gw = ok ? g[r] * inv : fixg[z * NREF + r];
        float qref = iz + refq[z * NREF + r] * scaleq;
        float zt = qok ? expf(3.f * (1.f - expf(gz * 2.f * (1.f - qref * invq))))
                       : expf(3.f);
        zeta[a * NREF + r] = zt * gw;
    }
}

// --------------------------------------- pass 3: dispersion (WMMA c6) -------
#define K3_BLOCK 128
#define K3_WAVES (K3_BLOCK / 32)

__global__ void __launch_bounds__(K3_BLOCK) k_disp(
    const float* __restrict__ r_ij, const int* __restrict__ idx_i,
    const int* __restrict__ idx_j, const int* __restrict__ Z,
    const float* __restrict__ zeta, const float* __restrict__ refc6,
    const float* __restrict__ sqrt_r4r2,
    const float* __restrict__ s6_raw, const float* __restrict__ s8_raw,
    const float* __restrict__ a1_raw, const float* __restrict__ a2_raw,
    float* __restrict__ y_atom, int P, int zmax) {
    // Per-wave staging:
    //   wt : A tile, w = vec(zi x zj) as f16, one 64-elem row per pair (lane)
    //   ctf: B tile, refc6 rows as raw f32 (filled by GLOBAL_LOAD_ASYNC_TO_LDS)
    //   dt : D tiles (16x16 f32 per group); diagonal = c6 per pair
    __shared__ _Float16 wt[K3_WAVES][32][64];
    __shared__ float    ctf[K3_WAVES][32][64];
    __shared__ float    dt[K3_WAVES][2][256];

    const int lane = threadIdx.x & 31;
    const int wv   = threadIdx.x >> 5;
    const int m    = lane & 15;   // pair row within group / matrix row
    const int hi   = lane >> 4;   // half-wave: selects contiguous K-subset (K 16*hi..16*hi+15)

    const float s6 = softplusf(s6_raw[0]);
    const float s8 = softplusf(s8_raw[0]);
    const float a1 = softplusf(a1_raw[0]);
    const float a2 = softplusf(a2_raw[0]);
    const float sq3 = 1.7320508075688772f;

    // Zero-pad K = 49..63 once; per-iteration writes never touch these bytes,
    // so both tiles stay exactly zero there (keeps the padded contraction exact
    // and avoids 0 x NaN from uninitialized LDS).
#pragma unroll
    for (int k2 = 49; k2 < 64; k2++) {
        wt[wv][lane][k2]  = (_Float16)0.f;
        ctf[wv][lane][k2] = 0.f;
    }

    // LDS byte offset of this lane's B-tile row (addrspace(3) offset lives in
    // the low 32 bits of the generic pointer).
    unsigned ldsrow = (unsigned)(uintptr_t)&ctf[wv][lane][0];

    int stride = gridDim.x * blockDim.x;
    int start  = blockIdx.x * blockDim.x + threadIdx.x;
    int iters  = (P + stride - 1) / stride;  // uniform across the grid

    for (int it = 0; it < iters; ++it) {
        int pr = start + it * stride;
        bool valid = pr < P;
        int p = valid ? pr : (P - 1);   // clamp: tail lanes process real pair P-1

        int i = idx_i[p], j = idx_j[p];
        int Zi = Z[i], Zj = Z[j];

        // Kick off the async global->LDS copy of this pair's refc6 row first
        // so it overlaps with all the VALU staging below.
        const float* crow = refc6 + ((Zi * zmax + Zj) * 49);
#pragma unroll
        for (int k2 = 0; k2 < 49; k2++) {
            asm volatile("global_load_async_to_lds_b32 %0, %1, off offset:%2"
                         :: "v"(ldsrow), "v"(crow), "i"(k2 * 4)
                         : "memory");
        }

        // Prefetch next iteration's r_ij chunk (speculative, linear stream).
        if (pr + stride < P)
            __builtin_prefetch(&r_ij[3 * (p + stride)], 0, 1);

        float rx = r_ij[3 * p], ry = r_ij[3 * p + 1], rz = r_ij[3 * p + 2];
        float d = sqrtf(rx * rx + ry * ry + rz * rz);

        float zi[7], zj[7];
#pragma unroll
        for (int r = 0; r < 7; r++) {
            zi[r] = zeta[i * 7 + r];
            zj[r] = zeta[j * 7 + r];
        }

        // Stage w = vec(zi x zj) (49 f16 values; padding already zeroed).
#pragma unroll
        for (int a = 0; a < 7; a++)
#pragma unroll
            for (int b = 0; b < 7; b++)
                wt[wv][lane][a * 7 + b] = (_Float16)(zi[a] * zj[b]);

        wait_asynccnt0();                 // B tile resident in LDS
        __builtin_amdgcn_wave_barrier();

        // Two pair-groups of 16, each contracted with two K=32 WMMA steps.
#pragma unroll
        for (int g2 = 0; g2 < 2; ++g2) {
            v8f acc = {};
#pragma unroll
            for (int q = 0; q < 2; q++) {
                v16h A, B;
#pragma unroll
                for (int e = 0; e < 16; e++) {
                    // A (16x32 f16): lanes 0-15 hold K 0-15, lanes 16-31 hold K 16-31
                    int ka = e + hi * 16 + 32 * q;
                    A[e] = wt[wv][g2 * 16 + m][ka];
                    // B (32x16): lanes 0-15 hold K 0-15, lanes 16-31 hold K 16-31
                    int kb = e + hi * 16 + 32 * q;
                    B[e] = (_Float16)ctf[wv][g2 * 16 + m][kb];
                }
                acc = __builtin_amdgcn_wmma_f32_16x16x32_f16(
                    false, A, false, B, (short)0, acc, false, false);
            }
            // D layout: elem v, lanes 0-15 -> (M=v, N=lane); lanes 16-31 -> (M=8+v)
#pragma unroll
            for (int v = 0; v < 8; v++)
                dt[wv][g2][(v + hi * 8) * 16 + m] = acc[v];
        }

        __builtin_amdgcn_wave_barrier();

        float c6 = dt[wv][hi][m * 16 + m];  // diagonal = this lane's pair

        // BJ damping with smooth cutoff shift.
        float r4 = sq3 * sqrt_r4r2[Zi] * sqrt_r4r2[Zj];
        float r0 = a1 * r4 + a2;
        float r02 = r0 * r0;
        float t6 = r02 * r02 * r02;
        float t8 = t6 * r02;
        const float cut6 = 177978.515625f;     // 7.5^6
        const float cut8 = 10011291.50390625f; // 7.5^8
        float c6t6 = cut6 + t6, c8t8 = cut8 + t8;
        float d2 = d * d;
        float d6 = d2 * d2 * d2;
        float d8 = d6 * d2;
        float lin = d / 7.5f - 1.f;
        float oor6 = 1.f / (d6 + t6) - 1.f / c6t6 + 6.f * cut6 / (c6t6 * c6t6) * lin;
        float oor8 = 1.f / (d8 + t8) - 1.f / c8t8 + 8.f * cut8 / (c8t8 * c8t8) * lin;
        if (d >= 7.5f) { oor6 = 0.f; oor8 = 0.f; }
        float edisp = -c6 * (s6 * oor6 + s8 * r4 * r4 * oor8) * 13.6056930121836215f;

        if (valid) atomicAdd(&y_atom[i], edisp);
    }
}

// -------------------------------------------- pass 4: molecule reduce -------
__global__ void k_mol(const float* __restrict__ y_atom, const int* __restrict__ idx_m,
                      float* __restrict__ y, int n) {
    int a = blockIdx.x * blockDim.x + threadIdx.x;
    if (a >= n) return;
    atomicAdd(&y[idx_m[a]], y_atom[a]);
}

// ---------------------------------------------------------------------------
extern "C" void kernel_launch(void* const* d_in, const int* in_sizes, int n_in,
                              void* d_out, int out_size, void* d_ws, size_t ws_size,
                              hipStream_t stream) {
    const float* r_ij       = (const float*)d_in[0];
    const float* qa         = (const float*)d_in[1];
    const float* s6_raw     = (const float*)d_in[2];
    const float* s8_raw     = (const float*)d_in[3];
    const float* a1_raw     = (const float*)d_in[4];
    const float* a2_raw     = (const float*)d_in[5];
    const float* scaleq_raw = (const float*)d_in[6];
    const float* rcov       = (const float*)d_in[7];
    const float* en         = (const float*)d_in[8];
    const float* zeff       = (const float*)d_in[9];
    const float* gam        = (const float*)d_in[10];
    const float* refq       = (const float*)d_in[11];
    const float* ncm        = (const float*)d_in[12];
    const float* ncw        = (const float*)d_in[13];
    const float* cn         = (const float*)d_in[14];
    const float* fixg       = (const float*)d_in[15];
    const float* sr4r2      = (const float*)d_in[16];
    const float* refc6      = (const float*)d_in[17];
    const int*   Z          = (const int*)d_in[18];
    const int*   idx_i      = (const int*)d_in[19];
    const int*   idx_j      = (const int*)d_in[20];
    const int*   idx_m      = (const int*)d_in[21];

    int N    = in_sizes[18];  // atoms
    int P    = in_sizes[19];  // pairs
    int zmax = in_sizes[7];   // ZMAX

    float* y      = (float*)d_out;
    float* covcn  = (float*)d_ws;        // [N]
    float* y_atom = covcn + N;           // [N]
    float* zeta   = y_atom + N;          // [N*7]
    (void)ws_size; (void)n_in;

    int mx = N > out_size ? N : out_size;
    k_init<<<(mx + 255) / 256, 256, 0, stream>>>(covcn, y_atom, y, N, out_size);
    k_cn<<<(P + 255) / 256, 256, 0, stream>>>(r_ij, idx_i, idx_j, Z, rcov, en,
                                              covcn, P);
    k_zeta<<<(N + 255) / 256, 256, 0, stream>>>(Z, covcn, qa, ncm, ncw, cn, fixg,
                                                refq, zeff, gam, scaleq_raw, zeta, N);
    k_disp<<<1024, K3_BLOCK, 0, stream>>>(r_ij, idx_i, idx_j, Z, zeta, refc6,
                                          sr4r2, s6_raw, s8_raw, a1_raw, a2_raw,
                                          y_atom, P, zmax);
    k_mol<<<(N + 255) / 256, 256, 0, stream>>>(y_atom, idx_m, y, N);
}
